// SWHSTransformerLayer_53171695125283
// MI455X (gfx1250) — matrix operand order repaired and checked by reading, not compiled, not run on hardware
//
#include <hip/hip_runtime.h>
#include <hip/hip_bf16.h>

// MI455X / gfx1250, wave32. WMMA f16->f32 accumulate for every matmul.
// Global->LDS staging uses gfx1250 async-to-LDS ops (ASYNCcnt) when available.
// Workspace requirement: ~750 MB carved from d_ws.

typedef __attribute__((ext_vector_type(16))) _Float16 v16h;
typedef __attribute__((ext_vector_type(8)))  float    v8f;
typedef __attribute__((ext_vector_type(4)))  int      v4i;
typedef __attribute__((ext_vector_type(2)))  int      v2i;

#define Bb   4
#define Ss   4096
#define Dd   1024
#define Hh   16
#define NSs  128
#define DAa  512
#define DFf  4096
#define DKk  32
#define DVv  64
#define KVW  1536          // DA + D
#define KVP  128           // padded kv_s row width (96 used, 128 stored)
#define NEGF (-3.0e38f)

#if __has_builtin(__builtin_amdgcn_global_load_async_to_lds_b128) && \
    __has_builtin(__builtin_amdgcn_global_load_async_to_lds_b64)
#define HAVE_ASYNC_LDS 1
#else
#define HAVE_ASYNC_LDS 0
#endif

#if HAVE_ASYNC_LDS
__device__ __forceinline__ void async_g2l_b128(const void* g, void* l) {
  __builtin_amdgcn_global_load_async_to_lds_b128(
      (__attribute__((address_space(1))) v4i*)g,
      (__attribute__((address_space(3))) v4i*)l, 0, 0);
}
__device__ __forceinline__ void async_g2l_b64(const void* g, void* l) {
  __builtin_amdgcn_global_load_async_to_lds_b64(
      (__attribute__((address_space(1))) v2i*)g,
      (__attribute__((address_space(3))) v2i*)l, 0, 0);
}
__device__ __forceinline__ void async_wait0() {
#if __has_builtin(__builtin_amdgcn_s_wait_asynccnt)
  __builtin_amdgcn_s_wait_asynccnt(0);
#else
  asm volatile("s_wait_asynccnt 0x0" ::: "memory");
#endif
}
#endif

// ---------------------------------------------------------------- cast f32->f16
__global__ __launch_bounds__(256) void cast_f16_kernel(const float* __restrict__ in,
                                                       _Float16* __restrict__ out,
                                                       long long n) {
  long long i = (long long)blockIdx.x * 256 + threadIdx.x;
  if (i < n) out[i] = (_Float16)in[i];
}

// ---------------------------------------------------------------- RMSNorm (one row of D=1024 per block)
__global__ __launch_bounds__(256) void rmsnorm_kernel(const float* __restrict__ x,
                                                      const float* __restrict__ w,
                                                      _Float16* __restrict__ outn,
                                                      _Float16* __restrict__ outraw) {
  __shared__ float red[256];
  size_t row = blockIdx.x;
  int tid = threadIdx.x;
  float loc[4];
  float ss = 0.f;
#pragma unroll
  for (int i = 0; i < 4; ++i) {
    loc[i] = x[row * Dd + tid + i * 256];
    ss += loc[i] * loc[i];
  }
  red[tid] = ss; __syncthreads();
  for (int st = 128; st > 0; st >>= 1) {
    if (tid < st) red[tid] += red[tid + st];
    __syncthreads();
  }
  float rinv = rsqrtf(red[0] * (1.0f / (float)Dd) + 1e-6f);
#pragma unroll
  for (int i = 0; i < 4; ++i) {
    int c = tid + i * 256;
    outn[row * Dd + c] = (_Float16)(loc[i] * rinv * w[c]);
    if (outraw) outraw[row * Dd + c] = (_Float16)loc[i];
  }
}

// ---------------------------------------------------------------- generic batched strided WMMA GEMM
// REQUIREMENTS: M % 32 == 0, N % 64 == 0, K % 32 == 0 (all call sites satisfy).
// C[z][m][n] = act( alpha * sum_k A[z][m][k] * B[z][k][n] + bias[n] ) (+ residual*rs)
// A addr: m*lda + k (K contiguous). B addr: k*bsk + n*bsn.
// batch offsets: off = (z/zdiv)*Out + (z%zdiv)*In
struct GemmP {
  const _Float16* A; long long lda, aOut, aIn;
  const _Float16* Bm; long long bsk, bsn, bOut, bIn;
  float* Cf; _Float16* Ch; long long cOut, cIn;
  const float* bias; const float* residual; const float* rs;
  int M, N, K, zdiv, act;   // act: 0 none, 1 gelu-exact
  float alpha;
};

#define BS_ST 72   // LDS B row stride in halves (144B, 16B multiple)
#define AS_ST 36   // LDS A row stride in halves (72B, 8B multiple)

__global__ __launch_bounds__(256) void gemm_wmma_kernel(GemmP p) {
  __shared__ _Float16 As[32 * AS_ST];   // 32 (M) x 32 (K)
  __shared__ _Float16 Bs[32 * BS_ST];   // 32 (K) x 64 (N)
  const int z = blockIdx.z;
  const long long zo = z / p.zdiv, zi = z % p.zdiv;
  const _Float16* __restrict__ A  = p.A  + zo * p.aOut + zi * p.aIn;
  const _Float16* __restrict__ Bm = p.Bm + zo * p.bOut + zi * p.bIn;
  const long long coff = zo * p.cOut + zi * p.cIn;

  const int tid = threadIdx.x, wave = tid >> 5, lane = tid & 31;
  const int m0 = blockIdx.y * 32, n0 = blockIdx.x * 64;
  const bool bRowMajor = (p.bsn == 1);

  v8f acc = {};
  for (int k0 = 0; k0 < p.K; k0 += 32) {
    __syncthreads();
    // ---- stage A tile (32x32): one 8B chunk per thread
    {
      int r = tid >> 3, c0 = (tid & 7) * 4;
      const _Float16* src = &A[(long long)(m0 + r) * p.lda + (k0 + c0)];
      _Float16* dst = &As[r * AS_ST + c0];
#if HAVE_ASYNC_LDS
      async_g2l_b64(src, dst);
#else
      *(uint2*)dst = *(const uint2*)src;
#endif
    }
    // ---- stage B tile (32x64)
    if (bRowMajor) {          // one 16B chunk per thread
      int rk = tid >> 3, n1 = (tid & 7) * 8;
      const _Float16* src = &Bm[(long long)(k0 + rk) * p.bsk + (n0 + n1)];
      _Float16* dst = &Bs[rk * BS_ST + n1];
#if HAVE_ASYNC_LDS
      async_g2l_b128(src, dst);
#else
      *(uint4*)dst = *(const uint4*)src;
#endif
    } else {                  // strided / transposed-B (only scores GEMM, K=32)
      int nn = tid >> 2, kk0 = (tid & 3) * 8;
#pragma unroll
      for (int i = 0; i < 8; ++i)
        Bs[(kk0 + i) * BS_ST + nn] =
            Bm[(long long)(k0 + kk0 + i) * p.bsk + (long long)(n0 + nn) * p.bsn];
    }
    // prefetch next K tile (speculative)
    if (k0 + 32 < p.K) {
      __builtin_prefetch(&A[(long long)(m0 + (tid >> 3)) * p.lda + k0 + 32], 0, 1);
      __builtin_prefetch(&Bm[(long long)(k0 + 32 + (tid >> 3)) * p.bsk + (long long)n0 * p.bsn], 0, 1);
    }
#if HAVE_ASYNC_LDS
    async_wait0();
#endif
    __syncthreads();

    // ---- fragments per ISA layout
    const int mL = (wave >> 2) * 16 + (lane & 15);
    const int kb = (lane >> 4) * 8;
    v16h av;
#pragma unroll
    for (int j = 0; j < 4; ++j) {
      av[2 * j]     = As[mL * AS_ST + kb + 2 * j];
      av[2 * j + 1] = As[mL * AS_ST + kb + 2 * j + 1];
      av[8 + 2 * j]     = As[mL * AS_ST + 16 + kb + 2 * j];
      av[8 + 2 * j + 1] = As[mL * AS_ST + 16 + kb + 2 * j + 1];
    }
    const int nL = (wave & 3) * 16 + (lane & 15);
    const int kbb = (lane >> 4) * 8;
    v16h bv;
#pragma unroll
    for (int j = 0; j < 8; ++j) {
      bv[j]     = Bs[(kbb + j) * BS_ST + nL];
      bv[8 + j] = Bs[(16 + kbb + j) * BS_ST + nL];
    }
    acc = __builtin_amdgcn_wmma_f32_16x16x32_f16(false, av, false, bv, (short)0, acc, false, false);
  }

  // ---- epilogue (C layout: lane half -> M 0-7 / 8-15, lane&15 -> N), no guards
#pragma unroll
  for (int r = 0; r < 8; ++r) {
    int m = m0 + (wave >> 2) * 16 + (lane >> 4) * 8 + r;
    int n = n0 + (wave & 3) * 16 + (lane & 15);
    float v = acc[r] * p.alpha;
    if (p.bias) v += p.bias[n];
    if (p.act == 1) v = 0.5f * v * (1.0f + erff(v * 0.70710678118f));
    long long idx = coff + (long long)m * p.N + n;
    if (p.residual) v = p.residual[idx] + v * p.rs[0];
    if (p.Cf) p.Cf[idx] = v;
    if (p.Ch) p.Ch[idx] = (_Float16)v;
  }
}

// ---------------------------------------------------------------- softmax over S (summary probs)
__global__ __launch_bounds__(256) void probs_softmax_kernel(const float* __restrict__ logits,
                                                            const float* __restrict__ mask,
                                                            _Float16* __restrict__ probs) {
  __shared__ float red[256];
  const int m = blockIdx.x, h = blockIdx.y, b = blockIdx.z;
  const int tid = threadIdx.x;
  float ev[16];
  float mx = NEGF;
#pragma unroll
  for (int i = 0; i < 16; ++i) {
    int s = tid + i * 256;
    float v = logits[((size_t)(b * Ss + s)) * (Hh * NSs) + h * NSs + m];
    v = (mask[b * Ss + s] > 0.f) ? v : NEGF;
    ev[i] = v;
    mx = fmaxf(mx, v);
  }
  red[tid] = mx; __syncthreads();
  for (int st = 128; st > 0; st >>= 1) { if (tid < st) red[tid] = fmaxf(red[tid], red[tid + st]); __syncthreads(); }
  mx = red[0]; __syncthreads();
  float sum = 0.f;
#pragma unroll
  for (int i = 0; i < 16; ++i) { ev[i] = expf(ev[i] - mx); sum += ev[i]; }
  red[tid] = sum; __syncthreads();
  for (int st = 128; st > 0; st >>= 1) { if (tid < st) red[tid] += red[tid + st]; __syncthreads(); }
  float inv = 1.0f / red[0];
  size_t base = (((size_t)(b * Hh + h)) * NSs + m) * Ss;
#pragma unroll
  for (int i = 0; i < 16; ++i) probs[base + tid + i * 256] = (_Float16)(ev[i] * inv);
}

// ---------------------------------------------------------------- softmax over NS=128 (summary attention)
__global__ __launch_bounds__(128) void att_softmax_kernel(const float* __restrict__ sc,
                                                          _Float16* __restrict__ attp) {
  __shared__ float red[128];
  size_t row = blockIdx.x;            // (b*H + h)*S + s
  int tid = threadIdx.x;
  float v = sc[row * NSs + tid];
  red[tid] = v; __syncthreads();
  for (int st = 64; st > 0; st >>= 1) { if (tid < st) red[tid] = fmaxf(red[tid], red[tid + st]); __syncthreads(); }
  float mx = red[0]; __syncthreads();
  float e = expf(v - mx);
  red[tid] = e; __syncthreads();
  for (int st = 64; st > 0; st >>= 1) { if (tid < st) red[tid] += red[tid + st]; __syncthreads(); }
  attp[row * NSs + tid] = (_Float16)(e / red[0]);
}

// ---------------------------------------------------------------- fused block-local sliding-window attention
// strides chosen so async B128 LDS destinations are 16B aligned
#define QST 40
#define KST 40
#define VST 72
#define PST 264
__global__ __launch_bounds__(256) void block_attn_kernel(const _Float16* __restrict__ qh,
                                                         const _Float16* __restrict__ kvh,
                                                         const float* __restrict__ out1,
                                                         _Float16* __restrict__ xatth) {
  extern __shared__ char smem[];
  _Float16* Qs = (_Float16*)smem;          // [128][QST]
  _Float16* Ks = Qs + 128 * QST;           // [256][KST]
  _Float16* Vs = Ks + 256 * KST;           // [256][VST]
  _Float16* Ps = Vs + 256 * VST;           // [128][PST]

  const int n = blockIdx.x, h = blockIdx.y, b = blockIdx.z;
  const int tid = threadIdx.x;
  const int sbase = n * NSs - NSs;         // ctx start (negative only for n==0)

#if HAVE_ASYNC_LDS
  {  // Q: 128x32, one B128 (8 halves) per thread
    int r = tid >> 1, c0 = (tid & 1) * 16;
    async_g2l_b128(&qh[((size_t)(b * Ss + n * NSs + r)) * DAa + h * DKk + c0],
                   &Qs[r * QST + c0]);
  }
  if (sbase >= 0) {
    // K: 256x32, two B128 per thread; V: 256x64, four B128 per thread
    const _Float16* kbase = &kvh[((size_t)(b * Ss + sbase + tid)) * KVW + h * 96];
#pragma unroll
    for (int c = 0; c < 32; c += 16)
      async_g2l_b128(kbase + c, &Ks[tid * KST + c]);
#pragma unroll
    for (int c = 0; c < 64; c += 16)
      async_g2l_b128(kbase + DKk + c, &Vs[tid * VST + c]);
    async_wait0();
  } else {
    async_wait0();
    for (int idx = tid; idx < 256 * 32; idx += 256) {
      int r = idx >> 5, c = idx & 31;
      int s = sbase + r;
      Ks[r * KST + c] = (s >= 0) ? kvh[((size_t)(b * Ss + s)) * KVW + h * 96 + c] : (_Float16)0.0f;
    }
    for (int idx = tid; idx < 256 * 64; idx += 256) {
      int r = idx >> 6, c = idx & 63;
      int s = sbase + r;
      Vs[r * VST + c] = (s >= 0) ? kvh[((size_t)(b * Ss + s)) * KVW + h * 96 + DKk + c] : (_Float16)0.0f;
    }
  }
#else
  for (int idx = tid; idx < 128 * 32; idx += 256) {
    int r = idx >> 5, c = idx & 31;
    Qs[r * QST + c] = qh[((size_t)(b * Ss + n * NSs + r)) * DAa + h * DKk + c];
  }
  for (int idx = tid; idx < 256 * 32; idx += 256) {
    int r = idx >> 5, c = idx & 31;
    int s = sbase + r;
    Ks[r * KST + c] = (s >= 0) ? kvh[((size_t)(b * Ss + s)) * KVW + h * 96 + c] : (_Float16)0.0f;
  }
  for (int idx = tid; idx < 256 * 64; idx += 256) {
    int r = idx >> 6, c = idx & 63;
    int s = sbase + r;
    Vs[r * VST + c] = (s >= 0) ? kvh[((size_t)(b * Ss + s)) * KVW + h * 96 + DKk + c] : (_Float16)0.0f;
  }
#endif
  __syncthreads();

  const int wave = tid >> 5, lane = tid & 31;
  const int mfrag = lane & 15, kb = (lane >> 4) * 8;
  const float scale = 0.17677669529663689f;   // 1/sqrt(32)

  v16h av;
  {
    int arow = wave * 16 + mfrag;
#pragma unroll
    for (int j = 0; j < 4; ++j) {
      av[2 * j]     = Qs[arow * QST + kb + 2 * j];
      av[2 * j + 1] = Qs[arow * QST + kb + 2 * j + 1];
      av[8 + 2 * j]     = Qs[arow * QST + 16 + kb + 2 * j];
      av[8 + 2 * j + 1] = Qs[arow * QST + 16 + kb + 2 * j + 1];
    }
  }
  for (int t = 0; t < 16; ++t) {
    v16h bv;
    int ncol = t * 16 + (lane & 15);
    int kbb = (lane >> 4) * 8;
#pragma unroll
    for (int j = 0; j < 8; ++j) {
      bv[j]     = Ks[ncol * KST + kbb + j];
      bv[8 + j] = Ks[ncol * KST + 16 + kbb + j];
    }
    v8f acc = {};
    acc = __builtin_amdgcn_wmma_f32_16x16x32_f16(false, av, false, bv, (short)0, acc, false, false);
#pragma unroll
    for (int r = 0; r < 8; ++r) {
      int a = wave * 16 + (lane >> 4) * 8 + r;
      int bcol = t * 16 + (lane & 15);
      bool valid = (bcol >= a) && (bcol <= a + NSs) && ((n > 0) || (bcol >= NSs));
      Ps[a * PST + bcol] = (_Float16)(valid ? acc[r] * scale : -60000.0f);
    }
  }
  __syncthreads();

  if (tid < 128) {
    float mx = NEGF;
    for (int j = 0; j < 256; ++j) mx = fmaxf(mx, (float)Ps[tid * PST + j]);
    float sum = 0.f;
    for (int j = 0; j < 256; ++j) sum += expf((float)Ps[tid * PST + j] - mx);
    float inv = 1.0f / sum;
    for (int j = 0; j < 256; ++j)
      Ps[tid * PST + j] = (_Float16)(expf((float)Ps[tid * PST + j] - mx) * inv);
  }
  __syncthreads();

  v8f acc4[4] = {};
  for (int ks = 0; ks < 8; ++ks) {
    v16h pa;
    int base = (wave * 16 + mfrag) * PST + ks * 32;
#pragma unroll
    for (int j = 0; j < 4; ++j) {
      pa[2 * j]     = Ps[base + kb + 2 * j];
      pa[2 * j + 1] = Ps[base + kb + 2 * j + 1];
      pa[8 + 2 * j]     = Ps[base + 16 + kb + 2 * j];
      pa[8 + 2 * j + 1] = Ps[base + 16 + kb + 2 * j + 1];
    }
#pragma unroll
    for (int nt = 0; nt < 4; ++nt) {
      v16h bv;
      int ncol = nt * 16 + (lane & 15);
      int kbb = ks * 32 + (lane >> 4) * 8;
#pragma unroll
      for (int j = 0; j < 8; ++j) {
        bv[j]     = Vs[(kbb + j) * VST + ncol];
        bv[8 + j] = Vs[(kbb + 16 + j) * VST + ncol];
      }
      acc4[nt] = __builtin_amdgcn_wmma_f32_16x16x32_f16(false, pa, false, bv, (short)0, acc4[nt], false, false);
    }
  }
#pragma unroll
  for (int nt = 0; nt < 4; ++nt) {
#pragma unroll
    for (int r = 0; r < 8; ++r) {
      int a = wave * 16 + (lane >> 4) * 8 + r;
      int d = nt * 16 + (lane & 15);
      int s = n * NSs + a;
      float o2 = (float)(_Float16)acc4[nt][r];   // reference casts out2 to f16
      float v = out1[(((size_t)(b * Hh + h)) * Ss + s) * DVv + d] + o2;
      xatth[((size_t)(b * Ss + s)) * Dd + h * DVv + d] = (_Float16)v;
    }
  }
}

// ================================================================ host
static inline char* ws_alloc(char*& cur, size_t bytes) {
  char* p = cur;
  cur += (bytes + 255) & ~(size_t)255;
  return p;
}

static void launch_gemm(hipStream_t st, const GemmP& p, int batches) {
  dim3 g((p.N + 63) / 64, (p.M + 31) / 32, batches);
  hipLaunchKernelGGL(gemm_wmma_kernel, g, dim3(256), 0, st, p);
}

extern "C" void kernel_launch(void* const* d_in, const int* in_sizes, int n_in,
                              void* d_out, int out_size, void* d_ws, size_t ws_size,
                              hipStream_t stream) {
  const float* x       = (const float*)d_in[0];
  const float* mask    = (const float*)d_in[1];
  const float* rs      = (const float*)d_in[2];
  const float* w_q     = (const float*)d_in[3];
  const float* b_q     = (const float*)d_in[4];
  const float* w_kv    = (const float*)d_in[5];
  const float* b_kv    = (const float*)d_in[6];
  const float* w_s     = (const float*)d_in[7];
  const float* b_s     = (const float*)d_in[8];
  const float* w_cat   = (const float*)d_in[9];
  const float* b_cat   = (const float*)d_in[10];
  const float* norm1_w = (const float*)d_in[11];
  const float* norm2_w = (const float*)d_in[12];
  const float* w1      = (const float*)d_in[13];
  const float* b1      = (const float*)d_in[14];
  const float* w2      = (const float*)d_in[15];
  const float* b2      = (const float*)d_in[16];
  float* out = (float*)d_out;

  const long long BS = (long long)Bb * Ss;           // 16384
  char* cur = (char*)d_ws;
  _Float16* xh    = (_Float16*)ws_alloc(cur, BS * Dd * 2);
  _Float16* xnh   = (_Float16*)ws_alloc(cur, BS * Dd * 2);
  _Float16* whq   = (_Float16*)ws_alloc(cur, (size_t)Dd * DAa * 2);
  _Float16* whkv  = (_Float16*)ws_alloc(cur, (size_t)Dd * KVW * 2);
  _Float16* whs   = (_Float16*)ws_alloc(cur, (size_t)Dd * (Hh * NSs) * 2);
  _Float16* whcat = (_Float16*)ws_alloc(cur, (size_t)Dd * Dd * 2);
  _Float16* wh1   = (_Float16*)ws_alloc(cur, (size_t)Dd * DFf * 2);
  _Float16* wh2   = (_Float16*)ws_alloc(cur, (size_t)DFf * Dd * 2);
  _Float16* qh    = (_Float16*)ws_alloc(cur, BS * DAa * 2);
  _Float16* kvh   = (_Float16*)ws_alloc(cur, BS * KVW * 2 + 512);   // +pad: kv_s B reads 128 cols
  float*    logits= (float*)   ws_alloc(cur, BS * (size_t)(Hh * NSs) * 4); // aliased as scores later
  _Float16* probs = (_Float16*)ws_alloc(cur, (size_t)Bb * Hh * NSs * Ss * 2);
  _Float16* kvsh  = (_Float16*)ws_alloc(cur, (size_t)Bb * Hh * NSs * KVP * 2);
  _Float16* attp  = (_Float16*)ws_alloc(cur, (size_t)Bb * Hh * Ss * NSs * 2);
  float*    out1  = (float*)   ws_alloc(cur, (size_t)Bb * Hh * Ss * DVv * 4);
  _Float16* xatth = (_Float16*)ws_alloc(cur, BS * Dd * 2);
  float*    xmid  = (float*)   ws_alloc(cur, BS * Dd * 4);
  _Float16* hh    = (_Float16*)ws_alloc(cur, BS * Dd * 2);
  _Float16* gh    = (_Float16*)ws_alloc(cur, BS * (size_t)DFf * 2);
  float* scores = logits;   // logits fully consumed before scores written

  // ---- cast weights to f16
  struct { const float* f; _Float16* h; long long n; } casts[6] = {
    { w_q,   whq,   (long long)Dd * DAa },
    { w_kv,  whkv,  (long long)Dd * KVW },
    { w_s,   whs,   (long long)Dd * Hh * NSs },
    { w_cat, whcat, (long long)Dd * Dd },
    { w1,    wh1,   (long long)Dd * DFf },
    { w2,    wh2,   (long long)DFf * Dd },
  };
  for (int i = 0; i < 6; ++i)
    hipLaunchKernelGGL(cast_f16_kernel, dim3((casts[i].n + 255) / 256), dim3(256), 0, stream,
                       casts[i].f, casts[i].h, casts[i].n);

  // ---- RMSNorm 1 (also writes raw f16 x)
  hipLaunchKernelGGL(rmsnorm_kernel, dim3((unsigned)BS), dim3(256), 0, stream, x, norm1_w, xnh, xh);

  GemmP p;

  // ---- q = xn @ w_q + b_q  -> f16
  p = {}; p.A = xnh; p.lda = Dd; p.Bm = whq; p.bsk = DAa; p.bsn = 1;
  p.Ch = qh; p.bias = b_q; p.M = (int)BS; p.N = DAa; p.K = Dd; p.zdiv = 1; p.alpha = 1.0f;
  launch_gemm(stream, p, 1);

  // ---- kv = xn @ w_kv + b_kv -> f16
  p = {}; p.A = xnh; p.lda = Dd; p.Bm = whkv; p.bsk = KVW; p.bsn = 1;
  p.Ch = kvh; p.bias = b_kv; p.M = (int)BS; p.N = KVW; p.K = Dd; p.zdiv = 1; p.alpha = 1.0f;
  launch_gemm(stream, p, 1);

  // ---- logits = x @ w_s + b_s -> f32 [B*S, H*NS]
  p = {}; p.A = xh; p.lda = Dd; p.Bm = whs; p.bsk = Hh * NSs; p.bsn = 1;
  p.Cf = logits; p.bias = b_s; p.M = (int)BS; p.N = Hh * NSs; p.K = Dd; p.zdiv = 1; p.alpha = 1.0f;
  launch_gemm(stream, p, 1);

  // ---- probs = softmax_s(logits with mask) -> f16 [b,h,m,s]
  hipLaunchKernelGGL(probs_softmax_kernel, dim3(NSs, Hh, Bb), dim3(256), 0, stream, logits, mask, probs);

  // ---- kv_s[b,h,m,0:96] = probs @ kv  (N padded to 128; cols 96..127 unused)
  p = {}; p.zdiv = Hh; p.alpha = 1.0f;
  p.A = probs; p.lda = Ss; p.aOut = (long long)Hh * NSs * Ss; p.aIn = (long long)NSs * Ss;
  p.Bm = kvh;  p.bsk = KVW; p.bsn = 1; p.bOut = (long long)Ss * KVW; p.bIn = 96;
  p.Ch = kvsh; p.cOut = (long long)Hh * NSs * KVP; p.cIn = (long long)NSs * KVP;
  p.M = NSs; p.N = KVP; p.K = Ss;
  launch_gemm(stream, p, Bb * Hh);

  // ---- scores = (q @ k_s^T) * 1/sqrt(DK) -> f32 [b,h,s,NS] (reuses logits buffer)
  p = {}; p.zdiv = Hh; p.alpha = 0.17677669529663689f;
  p.A = qh; p.lda = DAa; p.aOut = (long long)Ss * DAa; p.aIn = DKk;
  p.Bm = kvsh; p.bsk = 1; p.bsn = KVP; p.bOut = (long long)Hh * NSs * KVP; p.bIn = (long long)NSs * KVP;
  p.Cf = scores; p.cOut = (long long)Hh * Ss * NSs; p.cIn = (long long)Ss * NSs;
  p.M = Ss; p.N = NSs; p.K = DKk;
  launch_gemm(stream, p, Bb * Hh);

  // ---- att = softmax_k(scores) -> f16
  hipLaunchKernelGGL(att_softmax_kernel, dim3((unsigned)(Bb * Hh * Ss)), dim3(128), 0, stream, scores, attp);

  // ---- out1 = att @ v_s -> f32 [b,h,s,DV]
  p = {}; p.zdiv = Hh; p.alpha = 1.0f;
  p.A = attp; p.lda = NSs; p.aOut = (long long)Hh * Ss * NSs; p.aIn = (long long)Ss * NSs;
  p.Bm = kvsh + DKk; p.bsk = KVP; p.bsn = 1; p.bOut = (long long)Hh * NSs * KVP; p.bIn = (long long)NSs * KVP;
  p.Cf = out1; p.cOut = (long long)Hh * Ss * DVv; p.cIn = (long long)Ss * DVv;
  p.M = Ss; p.N = DVv; p.K = NSs;
  launch_gemm(stream, p, Bb * Hh);

  // ---- fused sliding-window block attention + combine -> x_att f16
  {
    size_t smem = (size_t)(128 * QST + 256 * KST + 256 * VST + 128 * PST) * sizeof(_Float16);
    (void)hipFuncSetAttribute((const void*)block_attn_kernel,
                              hipFuncAttributeMaxDynamicSharedMemorySize, (int)smem);
    hipLaunchKernelGGL(block_attn_kernel, dim3(Ss / NSs, Hh, Bb), dim3(256), smem, stream,
                       qh, kvh, out1, xatth);
  }

  // ---- x_mid = x + (x_att @ w_cat + b_cat) * rs -> f32
  p = {}; p.A = xatth; p.lda = Dd; p.Bm = whcat; p.bsk = Dd; p.bsn = 1;
  p.Cf = xmid; p.bias = b_cat; p.residual = x; p.rs = rs;
  p.M = (int)BS; p.N = Dd; p.K = Dd; p.zdiv = 1; p.alpha = 1.0f;
  launch_gemm(stream, p, 1);

  // ---- h = rms(x_mid) * norm2_w -> f16
  hipLaunchKernelGGL(rmsnorm_kernel, dim3((unsigned)BS), dim3(256), 0, stream, xmid, norm2_w, hh, (_Float16*)nullptr);

  // ---- g = gelu(h @ w1 + b1) -> f16
  p = {}; p.A = hh; p.lda = Dd; p.Bm = wh1; p.bsk = DFf; p.bsn = 1;
  p.Ch = gh; p.bias = b1; p.act = 1; p.M = (int)BS; p.N = DFf; p.K = Dd; p.zdiv = 1; p.alpha = 1.0f;
  launch_gemm(stream, p, 1);

  // ---- out = x_mid + (g @ w2 + b2) * rs -> f32 (final output)
  p = {}; p.A = gh; p.lda = DFf; p.Bm = wh2; p.bsk = Dd; p.bsn = 1;
  p.Cf = out; p.bias = b2; p.residual = xmid; p.rs = rs;
  p.M = (int)BS; p.N = Dd; p.K = DFf; p.zdiv = 1; p.alpha = 1.0f;
  launch_gemm(stream, p, 1);

  (void)in_sizes; (void)n_in; (void)out_size; (void)ws_size;
}
